// MinamoTopoModel_84834194031309
// MI455X (gfx1250) — matrix operand order repaired and checked by reading, not compiled, not run on hardware
//
#include <hip/hip_runtime.h>
#include <hip/hip_bf16.h>

// ---------------------------------------------------------------------------
// MI455X (gfx1250) GAT pipeline.
// Big GEMMs: bf16 WMMA (v_wmma_f32_16x16x32_bf16) with Tensor-Data-Mover
// (tensor_load_to_lds, TENSORcnt) double-buffered LDS staging.
// Edge softmax/scatter: native f32 atomics. Wave32 throughout.
// ---------------------------------------------------------------------------

typedef __attribute__((ext_vector_type(16))) __bf16 v16bf;
typedef __attribute__((ext_vector_type(8)))  float  v8f;
typedef __attribute__((ext_vector_type(4)))  unsigned int u32x4;
typedef __attribute__((ext_vector_type(4)))  int i32x4;
typedef __attribute__((ext_vector_type(8)))  int i32x8;

#if __has_builtin(__builtin_amdgcn_tensor_load_to_lds)
#define USE_TDM 1
#else
#define USE_TDM 0
#endif

#define N_NODES 20000
#define E_EDGES 100000
#define G_GRAPHS 16
#define EMB 128
#define HID 256
#define OUT_C 512
#define FEAT 512
#define HEADS 8
#define HC (HEADS * HID)            // 2048
#define ETOT (E_EDGES + N_NODES)    // 120000 (self-loops appended)
#define SLOPE 0.2f

static __device__ __forceinline__ float lrelu(float x) { return x > 0.f ? x : SLOPE * x; }
static __device__ __forceinline__ int imin(int a, int b) { return a < b ? a : b; }

// fp32 -> bf16 round-to-nearest-even, stored as raw u16
static __device__ __forceinline__ unsigned short f2bf(float f) {
  unsigned u = __float_as_uint(f);
  u += 0x7FFFu + ((u >> 16) & 1u);
  return (unsigned short)(u >> 16);
}

static __device__ __forceinline__ void atomicMaxF(float* addr, float v) {
  // monotonic int/uint mapping of IEEE floats
  if (v >= 0.f) atomicMax((int*)addr, __float_as_int(v));
  else          atomicMin((unsigned int*)addr, __float_as_uint(v));
}

// ---------------------------------------------------------------------------
// elementwise helpers
// ---------------------------------------------------------------------------
__global__ void k_cvt_bf16(const float* __restrict__ in, unsigned short* __restrict__ out, long n) {
  long i = (long)blockIdx.x * 256 + threadIdx.x;
  if (i < n) out[i] = f2bf(in[i]);
}

// weight convert + transpose: out[n*K + k] = bf16(in[k*Nc + n])  (Bt is [Nc,K])
__global__ void k_cvt_bf16_T(const float* __restrict__ in, unsigned short* __restrict__ out,
                             int K, int Nc) {
  long i = (long)blockIdx.x * 256 + threadIdx.x;
  long total = (long)K * Nc;
  if (i >= total) return;
  int n = (int)(i / K), k = (int)(i - (long)(i / K) * K);
  out[i] = f2bf(in[(long)k * Nc + n]);
}

__global__ void k_fill(float* __restrict__ p, float v, long n) {
  long i = (long)blockIdx.x * 256 + threadIdx.x;
  if (i < n) p[i] = v;
}

// out_f32 = lrelu(in + bias[col]) ; out_bf16 = bf16(out_f32)
__global__ void k_bias_act_cvt(const float* __restrict__ in, const float* __restrict__ bias,
                               float* __restrict__ outf, unsigned short* __restrict__ outb,
                               int nc, long total) {
  long i = (long)blockIdx.x * 256 + threadIdx.x;
  if (i >= total) return;
  float v = lrelu(in[i] + bias[(int)(i % nc)]);
  outf[i] = v;
  outb[i] = f2bf(v);
}

// ---------------------------------------------------------------------------
// TDM: issue a 2D tile load (tile_rows x 32 bf16 elements, row stride =
// row_stride_elems) from global into LDS at byte offset lds_off.
// Descriptor per CDNA5 ISA §8.3/8.4: group0 = {flags/count, lds_addr,
// global_addr, type=2}; group1 = {data_size, tensor dims, tile dims, strides}.
// ---------------------------------------------------------------------------
#if USE_TDM
static __device__ __forceinline__ void tdm_tile_load(
    unsigned lds_off, const unsigned short* g, unsigned tile_rows,
    unsigned long long row_stride_elems)
{
  unsigned long long ga = (unsigned long long)(uintptr_t)g;
  u32x4 g0 = { 1u,                                   // count=1 (valid), user mode
               lds_off,                              // lds_addr (bytes)
               (unsigned)ga,                         // global_addr[31:0]
               ((unsigned)(ga >> 32) & 0x01FFFFFFu) | 0x80000000u }; // [56:32] | type=2
  const unsigned td0 = 32u;          // tensor_dim0 (elements along K slab)
  const unsigned td1 = tile_rows;    // tensor_dim1
  i32x8 g1 = {
    (int)(1u << 16),                                            // data_size=1 -> 2 bytes
    (int)((td0 & 0xFFFFu) << 16),                               // [63:48] = td0 lo16
    (int)(((td0 >> 16) & 0xFFFFu) | ((td1 & 0xFFFFu) << 16)),   // [79:64] td0 hi, [95:80] td1 lo
    (int)(((td1 >> 16) & 0xFFFFu) | (32u << 16)),               // [111:96] td1 hi, [127:112] tile_dim0=32
    (int)(tile_rows & 0xFFFFu),                                 // [143:128] tile_dim1 (tile_dim2=0)
    (int)(row_stride_elems & 0xFFFFFFFFu),                      // tensor_dim0_stride lo32
    (int)((row_stride_elems >> 32) & 0xFFFFu),                  // stride hi16 (dim1_stride=0)
    0 };
  i32x4 z4 = {0, 0, 0, 0};
#if __clang_major__ >= 23
  i32x8 z8 = {0, 0, 0, 0, 0, 0, 0, 0};
  __builtin_amdgcn_tensor_load_to_lds(g0, g1, z4, z4, z8, 0);
#else
  __builtin_amdgcn_tensor_load_to_lds(g0, g1, z4, z4, 0);
#endif
}
#endif

// ---------------------------------------------------------------------------
// one 32-wide K step of WMMA on staged tiles (pA=[128 rows][32 k] bf16,
// pB=[128 cols][32 k] bf16). Each wave: 2x4 16x16 tiles of C.
// ---------------------------------------------------------------------------
static __device__ __forceinline__ void mma_step(
    const unsigned short* __restrict__ pA, const unsigned short* __restrict__ pB,
    int wm, int wn, int l16, int kh, v8f acc[2][4])
{
#pragma unroll
  for (int sm = 0; sm < 2; ++sm) {
    // A fragment (ISA 7.12.2, 16-bit A 16x32): lane<16 holds K 0..7,16..23;
    // lane>=16 holds K 8..15,24..31 of row M=l16. Pairs are K-contiguous.
    union { unsigned u[8]; v16bf v; } af;
    const int arow = (wm * 32 + sm * 16 + l16) * 32;
#pragma unroll
    for (int q = 0; q < 4; ++q) {
      int k = kh * 8 + 2 * q;
      af.u[q]     = *(const unsigned*)&pA[arow + k];
      af.u[4 + q] = *(const unsigned*)&pA[arow + 16 + k];
    }
#pragma unroll
    for (int sn = 0; sn < 4; ++sn) {
      // B fragment (16-bit B 32x16): VGPR q = rows K = kh*8 + 2q + (q>>2)*8, +1; N = l16.
      union { unsigned u[8]; v16bf v; } bfg;
      const int bcol = (wn * 64 + sn * 16 + l16) * 32;
#pragma unroll
      for (int q = 0; q < 8; ++q)
        bfg.u[q] = *(const unsigned*)&pB[bcol + kh * 8 + 2 * q + (q >> 2) * 8];

      acc[sm][sn] = __builtin_amdgcn_wmma_f32_16x16x32_bf16(
          false, af.v, false, bfg.v, (short)0, acc[sm][sn], false, false);
    }
  }
}

// ---------------------------------------------------------------------------
// bf16 WMMA GEMM: C[M,Nc] = act(A[M,K] @ Bt^T + bias),  Bt stored [Nc,K].
// block = 256 threads = 8 waves; block tile 128x128, K-step 32.
// TDM double-buffered LDS staging (wave 0 issues, s_wait_tensorcnt pipelining).
// ---------------------------------------------------------------------------
__global__ __launch_bounds__(256) void k_gemm_wmma(
    const unsigned short* __restrict__ A,    // [M,K]  bf16
    const unsigned short* __restrict__ Bt,   // [Nc,K] bf16 (pre-transposed weights)
    const float* __restrict__ bias,          // [Nc] or nullptr
    float* __restrict__ C,                   // [M,Nc] fp32
    int M, int K, int Nc, int act)
{
  const int tid  = threadIdx.x;
  const int wave = tid >> 5, lane = tid & 31;
  const int l16  = lane & 15, kh = lane >> 4;
  const int wm   = wave & 3,  wn = wave >> 2;

  const int bm0 = blockIdx.y * 128;
  const int bn0 = blockIdx.x * 128;

  v8f acc[2][4];
#pragma unroll
  for (int a = 0; a < 2; ++a)
#pragma unroll
    for (int b = 0; b < 4; ++b) acc[a][b] = (v8f){0.f, 0.f, 0.f, 0.f, 0.f, 0.f, 0.f, 0.f};

#if USE_TDM
  __shared__ unsigned short sA[2][128 * 32];
  __shared__ unsigned short sB[2][128 * 32];

  const int nk = K >> 5;
  const unsigned arows = (unsigned)imin(128, M - bm0);
  const unsigned ldsA0 = (unsigned)(uintptr_t)&sA[0][0];
  const unsigned ldsA1 = (unsigned)(uintptr_t)&sA[1][0];
  const unsigned ldsB0 = (unsigned)(uintptr_t)&sB[0][0];
  const unsigned ldsB1 = (unsigned)(uintptr_t)&sB[1][0];

  if (tid == 0) {  // wave 0 issues stage 0 (EXEC ignored by TDM; one issue per block)
    tdm_tile_load(ldsA0, A + (long)bm0 * K, arows, (unsigned long long)K);
    tdm_tile_load(ldsB0, Bt + (long)bn0 * K, 128u, (unsigned long long)K);
  }
  for (int i = 0; i < nk; ++i) {
    const int buf = i & 1;
    if (tid == 0) {
      if (i + 1 < nk) {   // prefetch next stage into the other buffer
        const long koff = (long)(i + 1) << 5;
        tdm_tile_load(buf ? ldsA0 : ldsA1, A + (long)bm0 * K + koff, arows,
                      (unsigned long long)K);
        tdm_tile_load(buf ? ldsB0 : ldsB1, Bt + (long)bn0 * K + koff, 128u,
                      (unsigned long long)K);
        __builtin_amdgcn_s_wait_tensorcnt(2);   // stage i complete, i+1 in flight
      } else {
        __builtin_amdgcn_s_wait_tensorcnt(0);
      }
    }
    __syncthreads();
    mma_step(sA[buf], sB[buf], wm, wn, l16, kh, acc);
    __syncthreads();
  }
#else
  __shared__ unsigned short sA1[128 * 32];   // [row][k]
  __shared__ unsigned short sB1[128 * 32];   // [n][k]
  for (int k0 = 0; k0 < K; k0 += 32) {
    for (int i = tid; i < 128 * 32; i += 256) {
      int r = i >> 5, c = i & 31;
      int gr = bm0 + r;
      sA1[i] = (gr < M) ? A[(long)gr * K + (k0 + c)] : (unsigned short)0;
    }
    for (int i = tid; i < 128 * 32; i += 256) {
      int nn = i >> 5, c = i & 31;
      sB1[i] = Bt[(long)(bn0 + nn) * K + (k0 + c)];
    }
    __syncthreads();
    mma_step(sA1, sB1, wm, wn, l16, kh, acc);
    __syncthreads();
  }
#endif

  // C/D layout: VGPR r -> lanes 0-15: M=r, lanes 16-31: M=r+8; N = l16.
#pragma unroll
  for (int sm = 0; sm < 2; ++sm)
#pragma unroll
    for (int sn = 0; sn < 4; ++sn)
#pragma unroll
      for (int r = 0; r < 8; ++r) {
        int grow = bm0 + wm * 32 + sm * 16 + kh * 8 + r;
        int gcol = bn0 + wn * 64 + sn * 16 + l16;
        if (grow < M) {
          float v = acc[sm][sn][r];
          if (bias) v += bias[gcol];
          if (act)  v = lrelu(v);
          C[(long)grow * Nc + gcol] = v;
        }
      }
}

// ---------------------------------------------------------------------------
// per-node attention logits: es[n,h] = sum_c h[n,h,c]*a_s[h,c] (same for ed)
// one block per node; one wave per head (wave32 shuffle reduction)
// ---------------------------------------------------------------------------
__global__ __launch_bounds__(256) void k_att_scores(
    const float* __restrict__ h, const float* __restrict__ a_s, const float* __restrict__ a_d,
    float* __restrict__ es, float* __restrict__ ed, int heads, int ch)
{
  const int n = blockIdx.x;
  const int wave = threadIdx.x >> 5, lane = threadIdx.x & 31;
  for (int hd = wave; hd < heads; hd += 8) {
    float ss = 0.f, sd = 0.f;
    const float* hp  = h   + (long)n * heads * ch + (long)hd * ch;
    const float* asp = a_s + (long)hd * ch;
    const float* adp = a_d + (long)hd * ch;
    for (int c = lane; c < ch; c += 32) {
      float x = hp[c];
      ss += x * asp[c];
      sd += x * adp[c];
    }
#pragma unroll
    for (int m = 16; m; m >>= 1) { ss += __shfl_xor(ss, m, 32); sd += __shfl_xor(sd, m, 32); }
    if (lane == 0) { es[(long)n * heads + hd] = ss; ed[(long)n * heads + hd] = sd; }
  }
}

// ---------------------------------------------------------------------------
// edge softmax, 3 passes (self-loops are edges e >= E_EDGES)
// ---------------------------------------------------------------------------
__global__ void k_edge_logits(const int* __restrict__ ei,
                              const float* __restrict__ es, const float* __restrict__ ed,
                              float* __restrict__ el, float* __restrict__ m, int heads)
{
  long t = (long)blockIdx.x * 256 + threadIdx.x;
  long tot = (long)ETOT * heads;
  if (t >= tot) return;
  int e = (int)(t / heads), hd = (int)(t - (long)(t / heads) * heads);
  int s, d;
  if (e < E_EDGES) { s = ei[e]; d = ei[E_EDGES + e]; } else { s = d = e - E_EDGES; }
  float v = lrelu(es[(long)s * heads + hd] + ed[(long)d * heads + hd]);
  el[t] = v;
  atomicMaxF(&m[(long)d * heads + hd], v);
}

__global__ void k_edge_exp(const int* __restrict__ ei, float* __restrict__ el,
                           const float* __restrict__ m, float* __restrict__ den, int heads)
{
  long t = (long)blockIdx.x * 256 + threadIdx.x;
  long tot = (long)ETOT * heads;
  if (t >= tot) return;
  int e = (int)(t / heads), hd = (int)(t - (long)(t / heads) * heads);
  int d = (e < E_EDGES) ? ei[E_EDGES + e] : (e - E_EDGES);
  float ex = __expf(el[t] - m[(long)d * heads + hd]);
  el[t] = ex;
  unsafeAtomicAdd(&den[(long)d * heads + hd], ex);
}

// one block per edge: out[dst] += h[src] * alpha (per-head)
__global__ __launch_bounds__(256) void k_edge_aggr(const int* __restrict__ ei,
    const float* __restrict__ ex, const float* __restrict__ den,
    const float* __restrict__ h, float* __restrict__ out, int heads, int ch)
{
  const int e = blockIdx.x;
  int s, d;
  if (e < E_EDGES) { s = ei[e]; d = ei[E_EDGES + e]; } else { s = d = e - E_EDGES; }
  __shared__ float alpha[8];
  if (threadIdx.x < heads)
    alpha[threadIdx.x] = ex[(long)e * heads + threadIdx.x] /
                         (den[(long)d * heads + threadIdx.x] + 1e-16f);
  __syncthreads();
  const int hc = heads * ch;
  const float* hs = h + (long)s * hc;
  float* od = out + (long)d * hc;
  for (int c = threadIdx.x; c < hc; c += 256)
    unsafeAtomicAdd(&od[c], hs[c] * alpha[c / ch]);
}

// ---------------------------------------------------------------------------
// global mean pool + head MLP
// ---------------------------------------------------------------------------
__global__ void k_pool_cnt(const int* __restrict__ batch, float* __restrict__ cnt) {
  int n = blockIdx.x * 256 + threadIdx.x;
  if (n < N_NODES) unsafeAtomicAdd(&cnt[batch[n]], 1.f);
}

__global__ __launch_bounds__(256) void k_pool_sum(const float* __restrict__ h,
                                                  const int* __restrict__ batch,
                                                  float* __restrict__ sums) {
  const int n = blockIdx.x;
  const int g = batch[n];
  for (int f = threadIdx.x; f < OUT_C; f += 256)
    unsafeAtomicAdd(&sums[(long)g * OUT_C + f], h[(long)n * OUT_C + f]);
}

__global__ __launch_bounds__(256) void k_head(const float* __restrict__ sums,
                                              const float* __restrict__ cnt,
                                              const float* __restrict__ Wf,
                                              const float* __restrict__ bf_,
                                              float* __restrict__ out) {
  __shared__ float mean[OUT_C];
  const int g = blockIdx.x;
  const float c = fmaxf(cnt[g], 1.f);
  for (int k = threadIdx.x; k < OUT_C; k += 256) mean[k] = sums[(long)g * OUT_C + k] / c;
  __syncthreads();
  for (int f = threadIdx.x; f < FEAT; f += 256) {
    float acc = bf_[f];
    for (int k = 0; k < OUT_C; ++k) acc += mean[k] * Wf[(long)k * FEAT + f];
    out[(long)g * FEAT + f] = lrelu(acc);
  }
}

// ---------------------------------------------------------------------------
// host orchestration
// ---------------------------------------------------------------------------
static inline size_t alignup(size_t x) { return (x + 255) & ~(size_t)255; }

extern "C" void kernel_launch(void* const* d_in, const int* in_sizes, int n_in,
                              void* d_out, int out_size, void* d_ws, size_t ws_size,
                              hipStream_t stream) {
  (void)in_sizes; (void)n_in; (void)out_size; (void)ws_size;
  const float* x   = (const float*)d_in[0];
  const int*   ei  = (const int*)d_in[1];
  const int*   bat = (const int*)d_in[2];
  const float* W0  = (const float*)d_in[3];  const float* b0 = (const float*)d_in[4];
  const float* W1  = (const float*)d_in[5];
  const float* a1s = (const float*)d_in[6];  const float* a1d = (const float*)d_in[7];
  const float* b1  = (const float*)d_in[8];
  const float* W2  = (const float*)d_in[9];
  const float* a2s = (const float*)d_in[10]; const float* a2d = (const float*)d_in[11];
  const float* b2  = (const float*)d_in[12];
  const float* W3  = (const float*)d_in[13];
  const float* a3s = (const float*)d_in[14]; const float* a3d = (const float*)d_in[15];
  const float* b3  = (const float*)d_in[16];
  const float* Wf  = (const float*)d_in[17]; const float* bfv = (const float*)d_in[18];
  float* out = (float*)d_out;

  // workspace carve-up
  char* ws = (char*)d_ws;
  size_t off = 0;
  auto take = [&](size_t bytes) { char* p = ws + off; off += alignup(bytes); return p; };

  unsigned short* xb   = (unsigned short*)take((size_t)N_NODES * 32 * 2);
  unsigned short* w0b  = (unsigned short*)take((size_t)32 * EMB * 2);      // [EMB,32]
  unsigned short* w1b  = (unsigned short*)take((size_t)EMB * HC * 2);      // [HC,EMB]
  unsigned short* w2b  = (unsigned short*)take((size_t)HC * HC * 2);       // [HC,HC]
  unsigned short* w3b  = (unsigned short*)take((size_t)HC * OUT_C * 2);    // [OUT,HC]
  unsigned short* hinb = (unsigned short*)take((size_t)N_NODES * HC * 2);  // bf16 activations
  float* hW   = (float*)take((size_t)N_NODES * HC * 4);                    // GEMM out / h for attn
  float* hagg = (float*)take((size_t)N_NODES * HC * 4);                    // aggregated
  float* es   = (float*)take((size_t)N_NODES * HEADS * 4);
  float* ed   = (float*)take((size_t)N_NODES * HEADS * 4);
  float* mbuf = (float*)take((size_t)N_NODES * HEADS * 4);
  float* den  = (float*)take((size_t)N_NODES * HEADS * 4);
  float* el   = (float*)take((size_t)ETOT * HEADS * 4);
  float* sums = (float*)take((size_t)G_GRAPHS * OUT_C * 4);
  float* cnt  = (float*)take((size_t)G_GRAPHS * 4);

  auto gblk = [](long n) { return (unsigned)((n + 255) / 256); };

  // x -> bf16 ; weights -> bf16 transposed [Nc,K] (TDM-friendly row-major tiles)
  k_cvt_bf16<<<gblk((long)N_NODES * 32), 256, 0, stream>>>(x, xb, (long)N_NODES * 32);
  k_cvt_bf16_T<<<gblk((long)32 * EMB), 256, 0, stream>>>(W0, w0b, 32, EMB);
  k_cvt_bf16_T<<<gblk((long)EMB * HC), 256, 0, stream>>>(W1, w1b, EMB, HC);
  k_cvt_bf16_T<<<gblk((long)HC * HC), 256, 0, stream>>>(W2, w2b, HC, HC);
  k_cvt_bf16_T<<<gblk((long)HC * OUT_C), 256, 0, stream>>>(W3, w3b, HC, OUT_C);

  const unsigned MB = (N_NODES + 127) / 128;  // 157

  // embed: h0 = lrelu(x@W0 + b0)  -> hW [N,EMB], then bf16 into hinb
  k_gemm_wmma<<<dim3(EMB / 128, MB), 256, 0, stream>>>(xb, w0b, b0, hW, N_NODES, 32, EMB, 1);
  k_cvt_bf16<<<gblk((long)N_NODES * EMB), 256, 0, stream>>>(hW, hinb, (long)N_NODES * EMB);

  struct Layer { const unsigned short* W; const float* as_; const float* ad_; const float* bias;
                 int K, heads, ch; };
  Layer layers[3] = {
    { w1b, a1s, a1d, b1, EMB, HEADS, HID },
    { w2b, a2s, a2d, b2, HC,  HEADS, HID },
    { w3b, a3s, a3d, b3, HC,  1,     OUT_C },
  };

  for (int l = 0; l < 3; ++l) {
    const Layer& L = layers[l];
    const int hc = L.heads * L.ch;

    // dense projection (no bias/act here — matches reference)
    k_gemm_wmma<<<dim3(hc / 128, MB), 256, 0, stream>>>(
        hinb, L.W, nullptr, hW, N_NODES, L.K, hc, 0);

    // attention logits per node
    k_att_scores<<<N_NODES, 256, 0, stream>>>(hW, L.as_, L.ad_, es, ed, L.heads, L.ch);

    // init reductions
    k_fill<<<gblk((long)N_NODES * L.heads), 256, 0, stream>>>(mbuf, -__builtin_huge_valf(),
                                                              (long)N_NODES * L.heads);
    k_fill<<<gblk((long)N_NODES * L.heads), 256, 0, stream>>>(den, 0.f, (long)N_NODES * L.heads);
    k_fill<<<gblk((long)N_NODES * hc), 256, 0, stream>>>(hagg, 0.f, (long)N_NODES * hc);

    // edge softmax + scatter aggregation
    k_edge_logits<<<gblk((long)ETOT * L.heads), 256, 0, stream>>>(ei, es, ed, el, mbuf, L.heads);
    k_edge_exp<<<gblk((long)ETOT * L.heads), 256, 0, stream>>>(ei, el, mbuf, den, L.heads);
    k_edge_aggr<<<ETOT, 256, 0, stream>>>(ei, el, den, hW, hagg, L.heads, L.ch);

    // out = lrelu(aggr + b); f32 stays in hagg, bf16 into hinb for next layer
    k_bias_act_cvt<<<gblk((long)N_NODES * hc), 256, 0, stream>>>(
        hagg, L.bias, hagg, hinb, hc, (long)N_NODES * hc);
  }

  // global mean pool (hagg is [N, OUT_C] after layer 3) + head MLP
  k_fill<<<gblk((long)G_GRAPHS * OUT_C), 256, 0, stream>>>(sums, 0.f, (long)G_GRAPHS * OUT_C);
  k_fill<<<1, 256, 0, stream>>>(cnt, 0.f, (long)G_GRAPHS);
  k_pool_cnt<<<gblk(N_NODES), 256, 0, stream>>>(bat, cnt);
  k_pool_sum<<<N_NODES, 256, 0, stream>>>(hagg, bat, sums);
  k_head<<<G_GRAPHS, 256, 0, stream>>>(sums, cnt, Wf, bfv, out);
}
